// SimCSE_25872882991576
// MI455X (gfx1250) — matrix operand order repaired and checked by reading, not compiled, not run on hardware
//
#include <hip/hip_runtime.h>
#include <hip/hip_bf16.h>
#include <stdint.h>

#define N_ROWS 8192
#define DDIM   1024
#define NSLICE 16
#define LOG2E  1.4426950408889634f
#define LN2    0.6931471805599453f

typedef __attribute__((ext_vector_type(16))) __bf16 v16bf;
typedef __attribute__((ext_vector_type(8)))  float  v8f;

union Frag { uint4 u[2]; v16bf v; };

__device__ inline uint16_t f32_to_bf16_rne(float f) {
    uint32_t u = __float_as_uint(f);
    uint32_t r = u + 0x7FFFu + ((u >> 16) & 1u);
    return (uint16_t)(r >> 16);
}
__device__ inline float bf16_to_f32(uint16_t h) {
    return __uint_as_float(((uint32_t)h) << 16);
}
__device__ inline float fast_exp2(float x) {
#if __has_builtin(__builtin_amdgcn_exp2f)
    return __builtin_amdgcn_exp2f(x);
#else
    return exp2f(x);
#endif
}
__device__ inline float fast_log2(float x) {
#if __has_builtin(__builtin_amdgcn_logf)
    return __builtin_amdgcn_logf(x);
#else
    return log2f(x);
#endif
}

// ---------------------------------------------------------------------------
// Kernel 1: split fp32 -> bf16 hi/lo.  rpr_r pre-scaled by log2(e) so the
// softmax path runs in base 2 (v_exp_f32 is exp2 natively).
// ---------------------------------------------------------------------------
__global__ void __launch_bounds__(256)
convert_split_kernel(const float* __restrict__ r, const float* __restrict__ l,
                     uint16_t* __restrict__ r_hi, uint16_t* __restrict__ r_lo,
                     uint16_t* __restrict__ l_hi, uint16_t* __restrict__ l_lo) {
    int i = blockIdx.x * blockDim.x + threadIdx.x;
    if (i >= N_ROWS * DDIM) return;
    float x = r[i] * LOG2E;
    uint16_t xh = f32_to_bf16_rne(x);
    r_hi[i] = xh;
    r_lo[i] = f32_to_bf16_rne(x - bf16_to_f32(xh));
    float y = l[i];
    uint16_t yh = f32_to_bf16_rne(y);
    l_hi[i] = yh;
    l_lo[i] = f32_to_bf16_rne(y - bf16_to_f32(yh));
}

// ---------------------------------------------------------------------------
// Kernel 2: fused split-bf16 WMMA GEMM + online logsumexp + diagonal grab.
// Wave tile: 32 rows x 32 cols (2 M-tiles x 2 N-tiles, 12 WMMA / 16 loads
// per K-step).  grid = (32 row-blocks, 16 col-slices), block = 256 (8 waves).
// ---------------------------------------------------------------------------
__global__ void __launch_bounds__(256)
gemm_lse_kernel(const uint16_t* __restrict__ r_hi, const uint16_t* __restrict__ r_lo,
                const uint16_t* __restrict__ l_hi, const uint16_t* __restrict__ l_lo,
                float2* __restrict__ ms_out,   // [N_ROWS][NSLICE] (m, s)
                float* __restrict__ diag_out)  // [N_ROWS] scaled diagonal
{
    const int lane = threadIdx.x & 31;
    const int wave = threadIdx.x >> 5;
    const int g    = lane >> 4;      // half-wave select
    const int lm   = lane & 15;
    const int i0   = (blockIdx.x * 8 + wave) * 32;
    const int jbase = blockIdx.y * (N_ROWS / NSLICE);   // 512-col slice

    // A fragments (ISA 16-bit A 16x32 layout: two contiguous 8-elem chunks
    // at d0 + g*8 and d0 + 16 + g*8).
    const uint16_t* a0h = r_hi + (size_t)(i0 + lm) * DDIM + g * 8;
    const uint16_t* a0l = r_lo + (size_t)(i0 + lm) * DDIM + g * 8;
    const uint16_t* a1h = a0h + (size_t)16 * DDIM;
    const uint16_t* a1l = a0l + (size_t)16 * DDIM;

    float m0[8], s0[8], m1[8], s1[8];
#pragma unroll
    for (int v = 0; v < 8; ++v) {
        m0[v] = m1[v] = -__builtin_inff();
        s0[v] = s1[v] = 0.f;
    }

    for (int jt = 0; jt < (N_ROWS / NSLICE) / 32; ++jt) {   // 16 iterations
        const int tc = jbase + jt * 32;                     // 32-col tile base
        // B fragments (same K order as A: 8-elem chunks at d0+g*8 and d0+16+g*8)
        const uint16_t* b0h = l_hi + (size_t)(tc + lm) * DDIM + g * 8;
        const uint16_t* b0l = l_lo + (size_t)(tc + lm) * DDIM + g * 8;
        const uint16_t* b1h = b0h + (size_t)16 * DDIM;
        const uint16_t* b1l = b0l + (size_t)16 * DDIM;

        v8f c00 = {0.f,0.f,0.f,0.f,0.f,0.f,0.f,0.f};
        v8f c01 = c00, c10 = c00, c11 = c00;

#pragma unroll 2
        for (int k = 0; k < DDIM; k += 32) {
            Frag fa0h, fa0l, fa1h, fa1l, fb0h, fb0l, fb1h, fb1l;
            fa0h.u[0] = *(const uint4*)(a0h + k);
            fa0h.u[1] = *(const uint4*)(a0h + k + 16);
            fa0l.u[0] = *(const uint4*)(a0l + k);
            fa0l.u[1] = *(const uint4*)(a0l + k + 16);
            fa1h.u[0] = *(const uint4*)(a1h + k);
            fa1h.u[1] = *(const uint4*)(a1h + k + 16);
            fa1l.u[0] = *(const uint4*)(a1l + k);
            fa1l.u[1] = *(const uint4*)(a1l + k + 16);
            fb0h.u[0] = *(const uint4*)(b0h + k);
            fb0h.u[1] = *(const uint4*)(b0h + k + 16);
            fb0l.u[0] = *(const uint4*)(b0l + k);
            fb0l.u[1] = *(const uint4*)(b0l + k + 16);
            fb1h.u[0] = *(const uint4*)(b1h + k);
            fb1h.u[1] = *(const uint4*)(b1h + k + 16);
            fb1l.u[0] = *(const uint4*)(b1l + k);
            fb1l.u[1] = *(const uint4*)(b1l + k + 16);

            // A*B ~= Ahi*Bhi + Alo*Bhi + Ahi*Blo   (split-bf16, 3 terms)
            c00 = __builtin_amdgcn_wmma_f32_16x16x32_bf16(false, fa0h.v, false, fb0h.v, (short)0, c00, false, false);
            c10 = __builtin_amdgcn_wmma_f32_16x16x32_bf16(false, fa1h.v, false, fb0h.v, (short)0, c10, false, false);
            c01 = __builtin_amdgcn_wmma_f32_16x16x32_bf16(false, fa0h.v, false, fb1h.v, (short)0, c01, false, false);
            c11 = __builtin_amdgcn_wmma_f32_16x16x32_bf16(false, fa1h.v, false, fb1h.v, (short)0, c11, false, false);
            c00 = __builtin_amdgcn_wmma_f32_16x16x32_bf16(false, fa0l.v, false, fb0h.v, (short)0, c00, false, false);
            c10 = __builtin_amdgcn_wmma_f32_16x16x32_bf16(false, fa1l.v, false, fb0h.v, (short)0, c10, false, false);
            c01 = __builtin_amdgcn_wmma_f32_16x16x32_bf16(false, fa0l.v, false, fb1h.v, (short)0, c01, false, false);
            c11 = __builtin_amdgcn_wmma_f32_16x16x32_bf16(false, fa1l.v, false, fb1h.v, (short)0, c11, false, false);
            c00 = __builtin_amdgcn_wmma_f32_16x16x32_bf16(false, fa0h.v, false, fb0l.v, (short)0, c00, false, false);
            c10 = __builtin_amdgcn_wmma_f32_16x16x32_bf16(false, fa1h.v, false, fb0l.v, (short)0, c10, false, false);
            c01 = __builtin_amdgcn_wmma_f32_16x16x32_bf16(false, fa0h.v, false, fb1l.v, (short)0, c01, false, false);
            c11 = __builtin_amdgcn_wmma_f32_16x16x32_bf16(false, fa1h.v, false, fb1l.v, (short)0, c11, false, false);
        }

        // Online softmax update (base 2).  C layout: row = v + 8g, col = lm.
        // A lane sees 2 columns (N-tile 0/1) of the same row -> one merge.
#pragma unroll
        for (int v = 0; v < 8; ++v) {
            float x0 = c00[v], x1 = c01[v];
            float mn = fmaxf(m0[v], fmaxf(x0, x1));
            s0[v] = s0[v] * fast_exp2(m0[v] - mn)
                  + fast_exp2(x0 - mn) + fast_exp2(x1 - mn);
            m0[v] = mn;

            float y0 = c10[v], y1 = c11[v];
            mn = fmaxf(m1[v], fmaxf(y0, y1));
            s1[v] = s1[v] * fast_exp2(m1[v] - mn)
                  + fast_exp2(y0 - mn) + fast_exp2(y1 - mn);
            m1[v] = mn;
        }

        // Diagonal tile: 32-aligned, occurs exactly when tc == i0 (uniform).
        if (tc == i0) {
#pragma unroll
            for (int v = 0; v < 8; ++v) {
                if (lm == v + 8 * g) {
                    diag_out[i0 + lm]      = c00[v];   // strip0 diag in N-tile0
                    diag_out[i0 + 16 + lm] = c11[v];   // strip1 diag in N-tile1
                }
            }
        }
    }

    // Butterfly merge of (m,s) across the 16 lanes sharing each row
    // (xor offsets < 16 never cross the half-wave boundary).
#pragma unroll
    for (int v = 0; v < 8; ++v) {
        float m = m0[v], s = s0[v];
#pragma unroll
        for (int off = 1; off < 16; off <<= 1) {
            float mo = __shfl_xor(m, off, 32);
            float so = __shfl_xor(s, off, 32);
            float mn = fmaxf(m, mo);
            s = s * fast_exp2(m - mn) + so * fast_exp2(mo - mn);
            m = mn;
        }
        if (lm == 0)
            ms_out[(size_t)(i0 + v + 8 * g) * NSLICE + blockIdx.y] = make_float2(m, s);

        m = m1[v]; s = s1[v];
#pragma unroll
        for (int off = 1; off < 16; off <<= 1) {
            float mo = __shfl_xor(m, off, 32);
            float so = __shfl_xor(s, off, 32);
            float mn = fmaxf(m, mo);
            s = s * fast_exp2(m - mn) + so * fast_exp2(mo - mn);
            m = mn;
        }
        if (lm == 0)
            ms_out[(size_t)(i0 + 16 + v + 8 * g) * NSLICE + blockIdx.y] = make_float2(m, s);
    }
}

// ---------------------------------------------------------------------------
// Kernel 3: merge NSLICE partials per row, per-row contribution
// (lse - diag)/N, block-reduce to one partial per block.
// ---------------------------------------------------------------------------
__global__ void __launch_bounds__(256)
row_reduce_kernel(const float2* __restrict__ ms, const float* __restrict__ diag,
                  float* __restrict__ partial) {
    __shared__ float sm[256];
    int r = blockIdx.x * 256 + threadIdx.x;
    float2 v[NSLICE];
    float m = -__builtin_inff();
#pragma unroll
    for (int s = 0; s < NSLICE; ++s) {
        v[s] = ms[(size_t)r * NSLICE + s];
        m = fmaxf(m, v[s].x);
    }
    float sum = 0.f;
#pragma unroll
    for (int s = 0; s < NSLICE; ++s) sum += v[s].y * fast_exp2(v[s].x - m);

    float lse_e  = (m + fast_log2(sum)) * LN2;   // back to natural log
    float diag_e = diag[r] * LN2;                // unscale diagonal
    float contrib = (lse_e - diag_e) * (1.0f / (float)N_ROWS);

    sm[threadIdx.x] = contrib;
    __syncthreads();
    for (int st = 128; st > 0; st >>= 1) {
        if (threadIdx.x < st) sm[threadIdx.x] += sm[threadIdx.x + st];
        __syncthreads();
    }
    if (threadIdx.x == 0) partial[blockIdx.x] = sm[0];
}

// ---------------------------------------------------------------------------
// Kernel 4: sum 32 block partials -> scalar output.
// ---------------------------------------------------------------------------
__global__ void __launch_bounds__(32)
final_kernel(const float* __restrict__ partial, float* __restrict__ out) {
    float x = partial[threadIdx.x];
#pragma unroll
    for (int off = 16; off > 0; off >>= 1) x += __shfl_down(x, off, 32);
    if (threadIdx.x == 0) out[0] = x;
}

extern "C" void kernel_launch(void* const* d_in, const int* in_sizes, int n_in,
                              void* d_out, int out_size, void* d_ws, size_t ws_size,
                              hipStream_t stream) {
    const float* rpr_r = (const float*)d_in[0];
    const float* rpr_l = (const float*)d_in[1];
    float* out = (float*)d_out;

    char* ws = (char*)d_ws;
    const size_t SZ = (size_t)N_ROWS * DDIM * sizeof(uint16_t);  // 16 MB each
    uint16_t* r_hi = (uint16_t*)(ws);
    uint16_t* r_lo = (uint16_t*)(ws + SZ);
    uint16_t* l_hi = (uint16_t*)(ws + 2 * SZ);
    uint16_t* l_lo = (uint16_t*)(ws + 3 * SZ);
    float2*   msb  = (float2*)(ws + 4 * SZ);
    float*    diag = (float*)(ws + 4 * SZ + (size_t)N_ROWS * NSLICE * sizeof(float2));
    float*    part = (float*)(ws + 4 * SZ + (size_t)N_ROWS * NSLICE * sizeof(float2)
                                 + (size_t)N_ROWS * sizeof(float));

    convert_split_kernel<<<(N_ROWS * DDIM + 255) / 256, 256, 0, stream>>>(
        rpr_r, rpr_l, r_hi, r_lo, l_hi, l_lo);

    gemm_lse_kernel<<<dim3(32, NSLICE), 256, 0, stream>>>(
        r_hi, r_lo, l_hi, l_lo, msb, diag);

    row_reduce_kernel<<<N_ROWS / 256, 256, 0, stream>>>(msb, diag, part);

    final_kernel<<<1, 32, 0, stream>>>(part, out);
}
